// GroupWiseLinearProjector_48601849921599
// MI455X (gfx1250) — matrix operand order repaired and checked by reading, not compiled, not run on hardware
//
#include <hip/hip_runtime.h>

// ---------------------------------------------------------------------------
// GroupWiseLinearProjector for MI455X (gfx1250, wave32, WMMA bf16).
// y[b,:,h,w] = Wg[g] @ x[b,:,h,w] + bg[g],  g = (h%4)*4 + (w%4)
// Split-precision bf16: x ~= xh+xl, w ~= wh+wl ; acc += xh*wh + xh*wl + xl*wh
// Fast path: weights pre-split into per-lane B-fragment layout in d_ws.
// ---------------------------------------------------------------------------

typedef __attribute__((ext_vector_type(16))) __bf16 v16bf;
typedef __attribute__((ext_vector_type(8)))  float  v8f;

#define NB    16
#define CIN   768
#define COUT  768
#define NH    64
#define NW    64
#define HW    (NH * NW)          // 4096
#define KSTEPS (CIN / 32)        // 24
#define NTILES (COUT / 16)       // 48
#define NT_CHUNK 4               // N-tiles (of 16) per accumulation chunk
#define LDS_ELEMS (NW * CIN)     // one bf16 plane: 64 pixels x 768 ch

// Pre-split W fragment pool: [g][ntile][ks][lane] -> 8 uints (32B)
#define NFRAGS   (16 * NTILES * KSTEPS * 32)          // 589824 fragments
#define FRAG_U4  ((size_t)NFRAGS * 2)                 // uint4s per plane
#define WS_BYTES ((size_t)NFRAGS * 32 * 2)            // hi + lo = 37,748,736 B

union Frag16 {
    uint4        q[2];
    unsigned int u[8];
    v16bf        v;
};

__device__ __forceinline__ unsigned int pack_bf16_pair(float f0, float f1) {
    __bf16 b0 = (__bf16)f0, b1 = (__bf16)f1;
    return (unsigned)__builtin_bit_cast(unsigned short, b0) |
           ((unsigned)__builtin_bit_cast(unsigned short, b1) << 16);
}

// ---------------------------------------------------------------------------
// Prep: split Wg (fp32) into bf16 hi/lo planes, stored in the exact per-lane
// WMMA B-fragment layout so the GEMM loop does pure coalesced b128 loads.
// One thread = one fragment-lane (32 B hi + 32 B lo).
// ---------------------------------------------------------------------------
__global__ __launch_bounds__(256) void wsplit_kernel(
    const float* __restrict__ Wg, uint4* __restrict__ wf)
{
    const unsigned idx = blockIdx.x * 256 + threadIdx.x;   // 0..NFRAGS-1
    const unsigned lane = idx & 31;
    const unsigned ks   = (idx >> 5) % KSTEPS;
    const unsigned nt   = (idx / (32 * KSTEPS)) % NTILES;
    const unsigned g    = idx / (32 * KSTEPS * NTILES);

    const unsigned o = nt * 16 + (lane & 15);
    const unsigned c = ks * 32 + 16 * (lane >> 4);
    const float* wrow = Wg + ((size_t)g * COUT + o) * CIN + c;

    const float4 w0 = ((const float4*)wrow)[0];
    const float4 w1 = ((const float4*)wrow)[1];
    const float4 w2 = ((const float4*)wrow)[2];
    const float4 w3 = ((const float4*)wrow)[3];
    const float fw[16] = {w0.x, w0.y, w0.z, w0.w, w1.x, w1.y, w1.z, w1.w,
                          w2.x, w2.y, w2.z, w2.w, w3.x, w3.y, w3.z, w3.w};

    Frag16 hi, lo;
#pragma unroll
    for (int e = 0; e < 8; ++e) {
        float  f0 = fw[2 * e], f1 = fw[2 * e + 1];
        __bf16 h0 = (__bf16)f0, h1 = (__bf16)f1;
        float  r0 = f0 - (float)h0, r1 = f1 - (float)h1;
        hi.u[e] = pack_bf16_pair(f0, f1);
        lo.u[e] = pack_bf16_pair(r0, r1);
    }
    wf[(size_t)idx * 2 + 0]           = hi.q[0];
    wf[(size_t)idx * 2 + 1]           = hi.q[1];
    wf[FRAG_U4 + (size_t)idx * 2 + 0] = lo.q[0];
    wf[FRAG_U4 + (size_t)idx * 2 + 1] = lo.q[1];
}

// ---------------------------------------------------------------------------
// Shared helper: stage row x[b,:,h,:] into LDS as swizzled bf16 hi/lo planes.
// ---------------------------------------------------------------------------
__device__ __forceinline__ void stage_x_row(
    const float* __restrict__ x, int b, int h, int tid,
    unsigned short* sh_hi, unsigned short* sh_lo)
{
    const size_t xbase = (size_t)b * ((size_t)CIN * HW) + (size_t)h * NW;
    for (int it = 0; it < 48; ++it) {            // 768*16 float4 / 256 thr
        int f  = it * 256 + tid;
        int c  = f >> 4;
        int w4 = f & 15;
        float4 v = *(const float4*)(x + xbase + (size_t)c * HW + w4 * 4);
        float vf[4] = {v.x, v.y, v.z, v.w};
        int cph = c ^ (w4 << 3);                 // w>>2 == w4 for w=4*w4+q
#pragma unroll
        for (int q = 0; q < 4; ++q) {
            int w = w4 * 4 + q;
            float  fv = vf[q];
            __bf16 hb = (__bf16)fv;
            float  rs = fv - (float)hb;
            __bf16 lb = (__bf16)rs;
            sh_hi[w * CIN + cph] = __builtin_bit_cast(unsigned short, hb);
            sh_lo[w * CIN + cph] = __builtin_bit_cast(unsigned short, lb);
        }
    }
}

// ---------------------------------------------------------------------------
// Fast GEMM: B fragments loaded pre-split from d_ws (pure b128 loads).
// ---------------------------------------------------------------------------
__global__ __launch_bounds__(256) void gwl_wmma_fast(
    const float* __restrict__ x, const uint4* __restrict__ wf,
    const float* __restrict__ bg, float* __restrict__ out)
{
    extern __shared__ unsigned short smem[];
    unsigned short* sh_hi = smem;
    unsigned short* sh_lo = smem + LDS_ELEMS;

    const int tid = threadIdx.x;
    const int bh  = blockIdx.x;
    const int b   = bh >> 6;
    const int h   = bh & 63;

    stage_x_row(x, b, h, tid, sh_hi, sh_lo);
    __syncthreads();

    const int lane = tid & 31;
    const int wave = tid >> 5;
    const int j    = wave & 3;
    const int ws   = wave >> 2;
    const int m    = lane & 15;
    const int half = lane >> 4;
    const int g    = (h & 3) * 4 + j;

    const unsigned rowbase = (unsigned)(4 * m + j) * CIN;
    const unsigned xorm    = (unsigned)(m << 3);
    const size_t   outb    = ((size_t)b * COUT) * HW + (size_t)h * NW;

    for (int q6 = 0; q6 < 6; ++q6) {
        const int nt0 = ws * 24 + q6 * NT_CHUNK;   // first N-tile of chunk
        const int nb  = nt0 * 16;

        v8f acc[NT_CHUNK];
#pragma unroll
        for (int t = 0; t < NT_CHUNK; ++t) acc[t] = (v8f){};

        for (int ks = 0; ks < KSTEPS; ++ks) {
            const int c0 = ks * 32;

            Frag16 ah, al;
            const unsigned bc0 = (unsigned)(c0 + 16 * half);
            const unsigned bc1 = bc0 + 8u;
            ah.q[0] = *(const uint4*)(sh_hi + rowbase + (bc0 ^ xorm));
            ah.q[1] = *(const uint4*)(sh_hi + rowbase + (bc1 ^ xorm));
            al.q[0] = *(const uint4*)(sh_lo + rowbase + (bc0 ^ xorm));
            al.q[1] = *(const uint4*)(sh_lo + rowbase + (bc1 ^ xorm));

            Frag16 bh_[NT_CHUNK], bl_[NT_CHUNK];
#pragma unroll
            for (int t = 0; t < NT_CHUNK; ++t) {
                const size_t fi =
                    (((size_t)g * NTILES + (nt0 + t)) * KSTEPS + ks) * 32 + lane;
                bh_[t].q[0] = wf[fi * 2 + 0];
                bh_[t].q[1] = wf[fi * 2 + 1];
                bl_[t].q[0] = wf[FRAG_U4 + fi * 2 + 0];
                bl_[t].q[1] = wf[FRAG_U4 + fi * 2 + 1];
            }
#pragma unroll
            for (int t = 0; t < NT_CHUNK; ++t) {
                acc[t] = __builtin_amdgcn_wmma_f32_16x16x32_bf16(
                    false, ah.v, false, bh_[t].v, (short)0, acc[t], false, false);
                acc[t] = __builtin_amdgcn_wmma_f32_16x16x32_bf16(
                    false, ah.v, false, bl_[t].v, (short)0, acc[t], false, false);
                acc[t] = __builtin_amdgcn_wmma_f32_16x16x32_bf16(
                    false, al.v, false, bh_[t].v, (short)0, acc[t], false, false);
            }
        }

#pragma unroll
        for (int t = 0; t < NT_CHUNK; ++t) {
            const int   o    = nb + t * 16 + m;
            const float bias = bg[g * COUT + o];
            float* orow = out + outb + (size_t)o * HW;
#pragma unroll
            for (int r = 0; r < 8; ++r) {
                const int w = 4 * (8 * half + r) + j;
                orow[w] = acc[t][r] + bias;
            }
        }
    }
}

// ---------------------------------------------------------------------------
// Fallback GEMM (d_ws too small): split W on the fly.
// ---------------------------------------------------------------------------
__global__ __launch_bounds__(256) void gwl_wmma_fallback(
    const float* __restrict__ x, const float* __restrict__ Wg,
    const float* __restrict__ bg, float* __restrict__ out)
{
    extern __shared__ unsigned short smem[];
    unsigned short* sh_hi = smem;
    unsigned short* sh_lo = smem + LDS_ELEMS;

    const int tid = threadIdx.x;
    const int bh  = blockIdx.x;
    const int b   = bh >> 6;
    const int h   = bh & 63;

    stage_x_row(x, b, h, tid, sh_hi, sh_lo);
    __syncthreads();

    const int lane = tid & 31;
    const int wave = tid >> 5;
    const int j    = wave & 3;
    const int ws   = wave >> 2;
    const int m    = lane & 15;
    const int half = lane >> 4;
    const int g    = (h & 3) * 4 + j;

    const unsigned rowbase = (unsigned)(4 * m + j) * CIN;
    const unsigned xorm    = (unsigned)(m << 3);
    const size_t   outb    = ((size_t)b * COUT) * HW + (size_t)h * NW;
    const float*   wgbase  = Wg + (size_t)g * COUT * CIN;

    for (int q6 = 0; q6 < 6; ++q6) {
        const int nb = ws * 384 + q6 * (NT_CHUNK * 16);

        v8f acc[NT_CHUNK];
#pragma unroll
        for (int t = 0; t < NT_CHUNK; ++t) acc[t] = (v8f){};

        for (int ks = 0; ks < KSTEPS; ++ks) {
            const int c0 = ks * 32;
            Frag16 ah, al;
            const unsigned bc0 = (unsigned)(c0 + 16 * half);
            const unsigned bc1 = bc0 + 8u;
            ah.q[0] = *(const uint4*)(sh_hi + rowbase + (bc0 ^ xorm));
            ah.q[1] = *(const uint4*)(sh_hi + rowbase + (bc1 ^ xorm));
            al.q[0] = *(const uint4*)(sh_lo + rowbase + (bc0 ^ xorm));
            al.q[1] = *(const uint4*)(sh_lo + rowbase + (bc1 ^ xorm));

#pragma unroll
            for (int t = 0; t < NT_CHUNK; ++t) {
                const int o = nb + t * 16 + m;
                const float* wrow =
                    wgbase + (size_t)o * CIN + (size_t)(c0 + 16 * half);
                const float4 w0 = ((const float4*)wrow)[0];
                const float4 w1 = ((const float4*)wrow)[1];
                const float4 w2 = ((const float4*)wrow)[2];
                const float4 w3 = ((const float4*)wrow)[3];
                const float fw[16] = {w0.x, w0.y, w0.z, w0.w,
                                      w1.x, w1.y, w1.z, w1.w,
                                      w2.x, w2.y, w2.z, w2.w,
                                      w3.x, w3.y, w3.z, w3.w};
                Frag16 bh_, bl_;
#pragma unroll
                for (int e = 0; e < 8; ++e) {
                    float  f0 = fw[2 * e], f1 = fw[2 * e + 1];
                    __bf16 h0 = (__bf16)f0, h1 = (__bf16)f1;
                    float  r0 = f0 - (float)h0, r1 = f1 - (float)h1;
                    bh_.u[e] = pack_bf16_pair(f0, f1);
                    bl_.u[e] = pack_bf16_pair(r0, r1);
                }
                acc[t] = __builtin_amdgcn_wmma_f32_16x16x32_bf16(
                    false, ah.v, false, bh_.v, (short)0, acc[t], false, false);
                acc[t] = __builtin_amdgcn_wmma_f32_16x16x32_bf16(
                    false, ah.v, false, bl_.v, (short)0, acc[t], false, false);
                acc[t] = __builtin_amdgcn_wmma_f32_16x16x32_bf16(
                    false, al.v, false, bh_.v, (short)0, acc[t], false, false);
            }
        }

#pragma unroll
        for (int t = 0; t < NT_CHUNK; ++t) {
            const int   o    = nb + t * 16 + m;
            const float bias = bg[g * COUT + o];
            float* orow = out + outb + (size_t)o * HW;
#pragma unroll
            for (int r = 0; r < 8; ++r) {
                const int w = 4 * (8 * half + r) + j;
                orow[w] = acc[t][r] + bias;
            }
        }
    }
}

extern "C" void kernel_launch(void* const* d_in, const int* in_sizes, int n_in,
                              void* d_out, int out_size, void* d_ws, size_t ws_size,
                              hipStream_t stream) {
    (void)in_sizes; (void)n_in; (void)out_size;
    const float* x  = (const float*)d_in[0];
    const float* Wg = (const float*)d_in[1];
    const float* bg = (const float*)d_in[2];
    float* out = (float*)d_out;

    const size_t shbytes = (size_t)2 * LDS_ELEMS * sizeof(unsigned short); // 192 KB

    if (ws_size >= WS_BYTES && d_ws != nullptr) {
        uint4* wf = (uint4*)d_ws;
        wsplit_kernel<<<dim3(NFRAGS / 256), dim3(256), 0, stream>>>(Wg, wf);
        (void)hipFuncSetAttribute((const void*)gwl_wmma_fast,
                                  hipFuncAttributeMaxDynamicSharedMemorySize,
                                  (int)shbytes);
        gwl_wmma_fast<<<dim3(NB * NH), dim3(256), shbytes, stream>>>(x, wf, bg, out);
    } else {
        (void)hipFuncSetAttribute((const void*)gwl_wmma_fallback,
                                  hipFuncAttributeMaxDynamicSharedMemorySize,
                                  (int)shbytes);
        gwl_wmma_fallback<<<dim3(NB * NH), dim3(256), shbytes, stream>>>(x, Wg, bg, out);
    }
}
